// ConvFlowNet_46806553592084
// MI455X (gfx1250) — compile-verified, matrix operand order checked by reading, not run on hardware
//
#include <hip/hip_runtime.h>
#include <hip/hip_bf16.h>

// ---------------------------------------------------------------------------
// FlowNetC forward, gfx1250. All convs = implicit GEMM on
// v_wmma_f32_16x16x32_bf16. Activations NHWC bf16 (fragment chunks =
// contiguous 8x bf16 -> global_load_b128). Weights pre-packed to bf16
// [Cout][Kpad64], K ordered (kh,kw,ci). A block = 8 waves sharing one
// 32-row M-panel; per 64-K round the block cooperatively stages the 32x64
// weight tile into LDS (double buffered, one barrier/round, prefetch of the
// next-next tile), then each wave runs 2 windows x 8 WMMAs (2M x 4N tile)
// with A from ds_load_b128 and B from guarded global b128 gathers.
// ---------------------------------------------------------------------------

typedef __attribute__((ext_vector_type(16))) __bf16          v16bf;
typedef __attribute__((ext_vector_type(8)))  float           v8f;
typedef __attribute__((ext_vector_type(8)))  unsigned short  ushort8;

union Frag { v16bf v; ushort8 h[2]; unsigned short u[16]; };

__device__ __forceinline__ unsigned short f2bf_bits(float f) {
  unsigned u = __float_as_uint(f);
  u += 0x7FFFu + ((u >> 16) & 1u);              // RNE
  return (unsigned short)(u >> 16);
}
__device__ __forceinline__ float bf2f(unsigned short h) {
  return __uint_as_float(((unsigned)h) << 16);
}

// ---------------------------------------------------------------------------
// conv + bias + LeakyReLU(0.1).  M = Cout, N = B*Hout*Wout,
// Kpad = roundup(KH*KW*Cin, 64), k = (kh*KW+kw)*Cin + ci.
// Cin = padded channel stride of input buffer (producers zero the padding).
// Requires Cout%32==0, N%64==0, Cin==8 or Cin%32==0 (all hold here).
// ---------------------------------------------------------------------------
template <bool OUT_F32>
__global__ __launch_bounds__(256) void conv_wmma_nhwc(
    const unsigned short* __restrict__ in, const unsigned short* __restrict__ wT,
    const float* __restrict__ bias, float* __restrict__ outF,
    unsigned short* __restrict__ outB,
    int Cin, int Hin, int Win, int Cout, int Hout, int Wout,
    int KH, int KW, int stride, int pad, int Kpad, int outC, int nBlocks) {
  __shared__ unsigned short smem[2][32 * 64];   // two 32x64 bf16 weight tiles

  const int lane  = threadIdx.x & 31;
  const int lcol  = lane & 15;
  const int khalf = lane >> 4;

  const int mBlocks = Cout >> 5;
  const int mb  = blockIdx.x % mBlocks;         // all 8 waves share this M-panel
  const int nbg = blockIdx.x / mBlocks;
  int nb = nbg * 8 + (threadIdx.x >> 5);
  const bool valid = nb < nBlocks;              // wave-uniform
  nb = valid ? nb : nBlocks - 1;                // clamp: keep barrier counts equal
  const int mBase = mb << 5;
  const int nBase = nb << 6;

  const int HW = Hout * Wout;
  const int G  = KH * KW;

  // Per-N-subtile pixel decomposition (this lane holds column lcol of each).
  int hi0A[4], wi0A[4], bA[4], hoA[4], woA[4], outPix[4];
  long long sBaseC[4];
#pragma unroll
  for (int j = 0; j < 4; ++j) {
    const int n  = nBase + (j << 4) + lcol;
    const int b  = n / HW;
    const int hw = n - b * HW;
    const int ho = hw / Wout;
    const int wo = hw - ho * Wout;
    bA[j] = b; hoA[j] = ho; woA[j] = wo;
    outPix[j] = (b * Hout + ho) * Wout + wo;
    hi0A[j] = ho * stride - pad;
    wi0A[j] = wo * stride - pad;
    sBaseC[j] = (((long long)b * Hin + hi0A[j]) * Win + wi0A[j]) * (long long)Cin;
  }

  // Lane's two 8-channel chunk offsets inside each 32-K window.
  const int relLo = khalf << 3;
  const int relHi = 16 + (khalf << 3);

  // Incremental (g, ci, kh, kw) chunk state; divisions only here.
  int gLo = relLo / Cin, cLo = relLo - gLo * Cin;
  int gHi = relHi / Cin, cHi = relHi - gHi * Cin;
  int khLo = gLo / KW, kwLo = gLo - khLo * KW;
  int khHi = gHi / KW, kwHi = gHi - khHi * KW;

  // Cooperative weight staging: thread -> (row, 8-col chunk) of 32x64 tile.
  const int srow = threadIdx.x >> 3;
  const int scol = (threadIdx.x & 7) << 3;
  const unsigned short* wSrc = wT + (size_t)(mBase + srow) * Kpad + scol;
  const int R = Kpad >> 6;                      // 64-K rounds

  auto stage = [&](int r, int bufI) {
    const unsigned short* s = wSrc + (size_t)r * 64;
    __builtin_prefetch((const void*)(s + 128), 0, 1);   // next-next round
    *(ushort8*)(&smem[bufI][srow * 64 + scol]) = *(const ushort8*)s;
  };

  v8f acc[2][4] = {};
  const ushort8 zz = {0, 0, 0, 0, 0, 0, 0, 0};

  stage(0, 0);
  __syncthreads();
  int curBuf = 0;

  for (int r = 0; r < R; ++r) {
    if (r + 1 < R) stage(r + 1, curBuf ^ 1);
    const unsigned short* sA = &smem[curBuf][0];

#pragma unroll
    for (int w = 0; w < 2; ++w) {               // two 32-K windows per round
      const int cbase = w << 5;
      Frag Af0, Af1;                            // A from LDS (ds_load_b128)
      Af0.h[0] = *(const ushort8*)(sA + lcol * 64 + cbase + relLo);
      Af0.h[1] = *(const ushort8*)(sA + lcol * 64 + cbase + relHi);
      Af1.h[0] = *(const ushort8*)(sA + (16 + lcol) * 64 + cbase + relLo);
      Af1.h[1] = *(const ushort8*)(sA + (16 + lcol) * 64 + cbase + relHi);

      const long long rowLo = ((long long)khLo * Win + kwLo) * (long long)Cin + cLo;
      const long long rowHi = ((long long)khHi * Win + kwHi) * (long long)Cin + cHi;
      const bool gokLo = gLo < G;
      const bool gokHi = gHi < G;

#pragma unroll
      for (int j = 0; j < 4; ++j) {
        Frag Bf;
        ushort8 t0 = zz, t1 = zz;
        const bool ok0 = gokLo &&
            (unsigned)(hi0A[j] + khLo) < (unsigned)Hin &&
            (unsigned)(wi0A[j] + kwLo) < (unsigned)Win;
        if (ok0) t0 = *(const ushort8*)(in + (sBaseC[j] + rowLo));
        const bool ok1 = gokHi &&
            (unsigned)(hi0A[j] + khHi) < (unsigned)Hin &&
            (unsigned)(wi0A[j] + kwHi) < (unsigned)Win;
        if (ok1) t1 = *(const ushort8*)(in + (sBaseC[j] + rowHi));
        Bf.h[0] = t0;
        Bf.h[1] = t1;

        acc[0][j] = __builtin_amdgcn_wmma_f32_16x16x32_bf16(
            false, Af0.v, false, Bf.v, (short)0, acc[0][j], false, false);
        acc[1][j] = __builtin_amdgcn_wmma_f32_16x16x32_bf16(
            false, Af1.v, false, Bf.v, (short)0, acc[1][j], false, false);
      }

      // Advance chunk K-state by 32 (branchless; layer-shape branch uniform).
      if (Cin >= 32) {
        cLo += 32;
        { const bool wr_ = cLo >= Cin; cLo = wr_ ? cLo - Cin : cLo; gLo += wr_; kwLo += wr_;
          const bool ww = kwLo >= KW; kwLo = ww ? 0 : kwLo; khLo += ww; }
        cHi += 32;
        { const bool wr_ = cHi >= Cin; cHi = wr_ ? cHi - Cin : cHi; gHi += wr_; kwHi += wr_;
          const bool ww = kwHi >= KW; kwHi = ww ? 0 : kwHi; khHi += ww; }
      } else {                                  // Cin == 8: 4 groups / window
        gLo += 4; kwLo += 4;
        { const bool ww = kwLo >= KW; kwLo = ww ? kwLo - KW : kwLo; khLo += ww; }
        gHi += 4; kwHi += 4;
        { const bool ww = kwHi >= KW; kwHi = ww ? kwHi - KW : kwHi; khHi += ww; }
      }
    }
    __syncthreads();                            // tile consumed; next tile staged
    curBuf ^= 1;
  }

  if (!valid) return;                           // duplicate waves skip the store

  // Epilogue: bias + LeakyReLU; NHWC bf16 stores are one b128 per fragment.
  const int mOff = khalf << 3;                  // D: lanes>=16 hold M = 8..15
#pragma unroll
  for (int mi = 0; mi < 2; ++mi) {
    const int ch0 = mBase + (mi << 4) + mOff;
#pragma unroll
    for (int j = 0; j < 4; ++j) {
      if (OUT_F32) {
#pragma unroll
        for (int r = 0; r < 8; ++r) {
          float v = acc[mi][j][r] + bias[ch0 + r];
          v = v > 0.0f ? v : v * 0.1f;
          outF[(((size_t)bA[j] * outC + ch0 + r) * Hout + hoA[j]) * Wout + woA[j]] = v;
        }
      } else {
        union { ushort8 h; unsigned short u[8]; } o;
#pragma unroll
        for (int r = 0; r < 8; ++r) {
          float v = acc[mi][j][r] + bias[ch0 + r];
          v = v > 0.0f ? v : v * 0.1f;
          o.u[r] = f2bf_bits(v);
        }
        *(ushort8*)(outB + (size_t)outPix[j] * outC + ch0) = o.h;
      }
    }
  }
}

// ---------------------------------------------------------------------------
// Weight pack: OIHW fp32 -> bf16 [Cout][Kpad], k = (kh*KW+kw)*Cpad + ci,
// zero beyond true Cin / true K.
// ---------------------------------------------------------------------------
__global__ __launch_bounds__(256) void pack_weights(
    const float* __restrict__ w, unsigned short* __restrict__ wT,
    int Cin, int Cpad, int KH, int KW, int Kpad, int total) {
  const int idx = blockIdx.x * blockDim.x + threadIdx.x;
  if (idx >= total) return;
  const int m  = idx / Kpad;
  const int k  = idx - m * Kpad;
  const int g  = k / Cpad;
  const int ci = k - g * Cpad;
  float v = 0.0f;
  if (g < KH * KW && ci < Cin) {
    const int kh = g / KW;
    const int kw = g - kh * KW;
    v = w[(((size_t)m * Cin + ci) * KH + kh) * KW + kw];
  }
  wT[idx] = f2bf_bits(v);
}

// ---------------------------------------------------------------------------
// Input pack: x [B,6,H,W] fp32 NCHW -> xa, xb NHWC bf16 [B,H,W,8] (ch 3..7 = 0)
// ---------------------------------------------------------------------------
__global__ __launch_bounds__(256) void pack_input(
    const float* __restrict__ x, unsigned short* __restrict__ xa,
    unsigned short* __restrict__ xb, int H, int W, int total) {
  const int p = blockIdx.x * blockDim.x + threadIdx.x;
  if (p >= total) return;
  const int w = p % W; int t = p / W;
  const int h = t % H; const int b = t / H;
  const size_t cs = (size_t)H * W;
  const size_t ib = (size_t)b * 6 * cs + (size_t)h * W + w;
  unsigned short* pa = xa + (size_t)p * 8;
  unsigned short* pb = xb + (size_t)p * 8;
#pragma unroll
  for (int c = 0; c < 8; ++c) {
    pa[c] = c < 3 ? f2bf_bits(x[ib + (size_t)c * cs]) : (unsigned short)0;
    pb[c] = c < 3 ? f2bf_bits(x[ib + (size_t)(c + 3) * cs]) : (unsigned short)0;
  }
}

// ---------------------------------------------------------------------------
// Cost volume (NHWC): 21x21 displacements, dilation 2. Contiguous channel
// dot via b128 loads; /C; LeakyReLU; writes cat channels [32, 473).
// ---------------------------------------------------------------------------
__global__ __launch_bounds__(256) void corr_kernel(
    const unsigned short* __restrict__ fa, const unsigned short* __restrict__ fb,
    unsigned short* __restrict__ cat, int C, int H, int W,
    int catC, int chanOff, int total) {
  const int idx = blockIdx.x * blockDim.x + threadIdx.x;
  if (idx >= total) return;
  const int wo = idx % W; int t = idx / W;
  const int ho = t % H;   t /= H;
  const int d  = t % 441;
  const int b  = t / 441;
  const int h2 = ho + (d / 21) * 2 - 20;
  const int w2 = wo + (d % 21) * 2 - 20;

  float acc = 0.0f;
  if ((unsigned)h2 < (unsigned)H && (unsigned)w2 < (unsigned)W) {
    const unsigned short* pa = fa + ((size_t)(b * H + ho) * W + wo) * C;
    const unsigned short* pb = fb + ((size_t)(b * H + h2) * W + w2) * C;
    for (int c0 = 0; c0 < C; c0 += 8) {
      union { ushort8 h; unsigned short u[8]; } a8, b8;
      a8.h = *(const ushort8*)(pa + c0);
      b8.h = *(const ushort8*)(pb + c0);
#pragma unroll
      for (int r = 0; r < 8; ++r) acc += bf2f(a8.u[r]) * bf2f(b8.u[r]);
    }
  }
  acc *= (1.0f / (float)C);
  acc = acc > 0.0f ? acc : acc * 0.1f;
  cat[((size_t)(b * H + ho) * W + wo) * catC + chanOff + d] = f2bf_bits(acc);
}

// Zero the concat buffer's padding channels [c0, catC).
__global__ __launch_bounds__(256) void pad_cat(
    unsigned short* __restrict__ cat, int catC, int c0, int total) {
  const int idx = blockIdx.x * blockDim.x + threadIdx.x;
  if (idx >= total) return;
  const int npad = catC - c0;
  const int pix = idx / npad;
  const int c   = idx - pix * npad;
  cat[(size_t)pix * catC + c0 + c] = 0;
}

// ---------------------------------------------------------------------------
// Host side
// ---------------------------------------------------------------------------
static inline int round64i(int v) { return (v + 63) & ~63; }

template <bool OUTF>
static void conv_launch(const unsigned short* in, const unsigned short* wT,
                        const float* bias, float* outF, unsigned short* outB,
                        int Bn, int Cin, int Hin, int Win,
                        int Cout, int Hout, int Wout,
                        int KH, int KW, int stride, int pad,
                        int outC, hipStream_t s) {
  const int Kpad    = round64i(KH * KW * Cin);
  const int N       = Bn * Hout * Wout;
  const int nBlocks = N / 64;
  const int mBlocks = Cout / 32;
  const int nbg     = (nBlocks + 7) / 8;
  dim3 grid(mBlocks * nbg), block(256);
  conv_wmma_nhwc<OUTF><<<grid, block, 0, s>>>(
      in, wT, bias, outF, outB, Cin, Hin, Win, Cout, Hout, Wout,
      KH, KW, stride, pad, Kpad, outC, nBlocks);
}

static void pack_w(const float* w, unsigned short* wT, int Cin, int Cpad,
                   int KH, int KW, int Cout, hipStream_t s) {
  const int Kpad  = round64i(KH * KW * Cpad);
  const int total = Cout * Kpad;
  pack_weights<<<(total + 255) / 256, 256, 0, s>>>(w, wT, Cin, Cpad, KH, KW, Kpad, total);
}

extern "C" void kernel_launch(void* const* d_in, const int* in_sizes, int n_in,
                              void* d_out, int out_size, void* d_ws, size_t ws_size,
                              hipStream_t stream) {
  (void)in_sizes; (void)n_in; (void)out_size; (void)ws_size;

  const float* x    = (const float*)d_in[0];
  const float* w1   = (const float*)d_in[1];  const float* b1   = (const float*)d_in[2];
  const float* w2   = (const float*)d_in[3];  const float* b2   = (const float*)d_in[4];
  const float* w3   = (const float*)d_in[5];  const float* b3   = (const float*)d_in[6];
  const float* wr   = (const float*)d_in[7];  const float* br   = (const float*)d_in[8];
  const float* w3_1 = (const float*)d_in[9];  const float* b3_1 = (const float*)d_in[10];
  const float* w4   = (const float*)d_in[11]; const float* b4   = (const float*)d_in[12];
  const float* w4_1 = (const float*)d_in[13]; const float* b4_1 = (const float*)d_in[14];
  const float* w5   = (const float*)d_in[15]; const float* b5   = (const float*)d_in[16];
  const float* w5_1 = (const float*)d_in[17]; const float* b5_1 = (const float*)d_in[18];
  const float* w6   = (const float*)d_in[19]; const float* b6   = (const float*)d_in[20];
  const float* w6_1 = (const float*)d_in[21]; const float* b6_1 = (const float*)d_in[22];
  float* out = (float*)d_out;

  const int Bn = 8;

  // ---- workspace carve (all bf16-as-ushort) ------------------------------
  char* p = (char*)d_ws;
  auto carve = [&](size_t elems) {
    char* r = p;
    p += (elems * 2 + 255) & ~(size_t)255;
    return (unsigned short*)r;
  };
  unsigned short* wt1   = carve((size_t)64   * round64i(49 * 8));
  unsigned short* wt2   = carve((size_t)128  * round64i(25 * 64));
  unsigned short* wt3   = carve((size_t)256  * round64i(25 * 128));
  unsigned short* wtr   = carve((size_t)32   * round64i(1 * 256));
  unsigned short* wt3_1 = carve((size_t)256  * round64i(9 * 480));
  unsigned short* wt4   = carve((size_t)512  * round64i(9 * 256));
  unsigned short* wt4_1 = carve((size_t)512  * round64i(9 * 512));
  unsigned short* wt5   = carve((size_t)512  * round64i(9 * 512));
  unsigned short* wt5_1 = carve((size_t)512  * round64i(9 * 512));
  unsigned short* wt6   = carve((size_t)1024 * round64i(9 * 512));
  unsigned short* wt6_1 = carve((size_t)1024 * round64i(9 * 1024));
  unsigned short* xa    = carve((size_t)Bn * 384 * 512 * 8);   // NHWC, ch 3->8 pad
  unsigned short* xb    = carve((size_t)Bn * 384 * 512 * 8);
  unsigned short* buf1  = carve((size_t)Bn * 192 * 256 * 64);
  unsigned short* buf2  = carve((size_t)Bn * 96  * 128 * 128);
  unsigned short* fa    = carve((size_t)Bn * 48  * 64  * 256);
  unsigned short* fb    = carve((size_t)Bn * 48  * 64  * 256);
  // Aliases over dead regions (timeline-safe; rewritten every launch).
  unsigned short* cat = xa;     // [B,48,64,480]  (11.8M <= 12.6M)
  unsigned short* pA  = xb;     // up to [B,48,64,256]
  unsigned short* pB  = buf1;   // up to [B,24,32,512]

  // ---- weight + input packing --------------------------------------------
  pack_w(w1,   wt1,   3,    8,    7, 7, 64,   stream);
  pack_w(w2,   wt2,   64,   64,   5, 5, 128,  stream);
  pack_w(w3,   wt3,   128,  128,  5, 5, 256,  stream);
  pack_w(wr,   wtr,   256,  256,  1, 1, 32,   stream);
  pack_w(w3_1, wt3_1, 473,  480,  3, 3, 256,  stream);
  pack_w(w4,   wt4,   256,  256,  3, 3, 512,  stream);
  pack_w(w4_1, wt4_1, 512,  512,  3, 3, 512,  stream);
  pack_w(w5,   wt5,   512,  512,  3, 3, 512,  stream);
  pack_w(w5_1, wt5_1, 512,  512,  3, 3, 512,  stream);
  pack_w(w6,   wt6,   512,  512,  3, 3, 1024, stream);
  pack_w(w6_1, wt6_1, 1024, 1024, 3, 3, 1024, stream);
  {
    const int total = Bn * 384 * 512;
    pack_input<<<(total + 255) / 256, 256, 0, stream>>>(x, xa, xb, 384, 512, total);
  }

  // ---- siamese towers (shared weights) -----------------------------------
  conv_launch<false>(xa, wt1, b1, nullptr, buf1, Bn, 8, 384, 512,
                     64, 192, 256, 7, 7, 2, 3, 64, stream);
  conv_launch<false>(buf1, wt2, b2, nullptr, buf2, Bn, 64, 192, 256,
                     128, 96, 128, 5, 5, 2, 2, 128, stream);
  conv_launch<false>(buf2, wt3, b3, nullptr, fa, Bn, 128, 96, 128,
                     256, 48, 64, 5, 5, 2, 2, 256, stream);
  conv_launch<false>(xb, wt1, b1, nullptr, buf1, Bn, 8, 384, 512,
                     64, 192, 256, 7, 7, 2, 3, 64, stream);
  conv_launch<false>(buf1, wt2, b2, nullptr, buf2, Bn, 64, 192, 256,
                     128, 96, 128, 5, 5, 2, 2, 128, stream);
  conv_launch<false>(buf2, wt3, b3, nullptr, fb, Bn, 128, 96, 128,
                     256, 48, 64, 5, 5, 2, 2, 256, stream);

  // ---- redirect 1x1 -> cat[:, :, :, 0:32] --------------------------------
  conv_launch<false>(fa, wtr, br, nullptr, cat, Bn, 256, 48, 64,
                     32, 48, 64, 1, 1, 1, 0, /*outC*/480, stream);

  // ---- correlation -> cat[:, :, :, 32:473]; zero pad 473:480 -------------
  {
    const int total = Bn * 441 * 48 * 64;
    corr_kernel<<<(total + 255) / 256, 256, 0, stream>>>(fa, fb, cat, 256,
                                                         48, 64, 480, 32, total);
    const int padTot = Bn * 48 * 64 * 7;
    pad_cat<<<(padTot + 255) / 256, 256, 0, stream>>>(cat, 480, 473, padTot);
  }

  // ---- fused trunk --------------------------------------------------------
  conv_launch<false>(cat, wt3_1, b3_1, nullptr, pA, Bn, 480, 48, 64,
                     256, 48, 64, 3, 3, 1, 1, 256, stream);
  conv_launch<false>(pA, wt4, b4, nullptr, pB, Bn, 256, 48, 64,
                     512, 24, 32, 3, 3, 2, 1, 512, stream);
  conv_launch<false>(pB, wt4_1, b4_1, nullptr, pA, Bn, 512, 24, 32,
                     512, 24, 32, 3, 3, 1, 1, 512, stream);
  conv_launch<false>(pA, wt5, b5, nullptr, pB, Bn, 512, 24, 32,
                     512, 12, 16, 3, 3, 2, 1, 512, stream);
  conv_launch<false>(pB, wt5_1, b5_1, nullptr, pA, Bn, 512, 12, 16,
                     512, 12, 16, 3, 3, 1, 1, 512, stream);
  conv_launch<false>(pA, wt6, b6, nullptr, pB, Bn, 512, 12, 16,
                     1024, 6, 8, 3, 3, 2, 1, 1024, stream);
  // final layer: fp32 NCHW straight to d_out
  conv_launch<true>(pB, wt6_1, b6_1, out, nullptr, Bn, 1024, 6, 8,
                    1024, 6, 8, 3, 3, 1, 1, 1024, stream);
}
